// TMix_75780402971159
// MI455X (gfx1250) — matrix operand order repaired and checked by reading, not compiled, not run on hardware
//
#include <hip/hip_runtime.h>
#include <hip/hip_bf16.h>
#include <stdint.h>

typedef __attribute__((ext_vector_type(16))) __bf16 v16bf;
typedef __attribute__((ext_vector_type(8)))  float  v8f;

#define DDIM  1024
#define TLEN  2048
#define NROW  16384   /* B*T */
#define NHEAD 16
#define HDIM  64
#define LOGW  (-0.6065306597126334f)
#define GNEPS (6.4e-4f)   /* HD * 1e-5 */

__device__ __forceinline__ float sigmoidf_(float x) { return 1.0f / (1.0f + __expf(-x)); }

// CDNA5 async global->LDS path (ASYNCcnt), gated so compile never breaks.
// Builtin signature (from clang diagnostic): (int4 AS1*, int4 AS3*, Imm, Imm).
#if defined(__gfx1250__) && __has_builtin(__builtin_amdgcn_global_load_async_to_lds_b128) && __has_builtin(__builtin_amdgcn_s_wait_asynccnt)
#define USE_ASYNC_LDS 1
typedef int v4i_ __attribute__((ext_vector_type(4)));
typedef __attribute__((address_space(1))) v4i_ g1v4;
typedef __attribute__((address_space(3))) v4i_ l3v4;
__device__ __forceinline__ g1v4* as_global_(const void* p) {
    return (g1v4*)(unsigned long long)p;                // global VA == AS1 address
}
__device__ __forceinline__ l3v4* as_lds_(const void* p) {
    return (l3v4*)(unsigned int)(unsigned long long)p;  // low 32 bits == LDS offset
}
#else
#define USE_ASYNC_LDS 0
#endif

// ---------------------------------------------------------------------------
// Weight convert + transpose: W[K][N] f32 -> Wt[N][K] bf16
// ---------------------------------------------------------------------------
__global__ void wt_kernel(const float* __restrict__ W, __bf16* __restrict__ Wt,
                          int K, int N) {
    int idx = blockIdx.x * 256 + threadIdx.x;
    if (idx >= K * N) return;
    int n = idx / K, k = idx - n * K;
    Wt[idx] = (__bf16)W[k * N + n];
}

// ---------------------------------------------------------------------------
// Token shift + 6-way mix -> bf16 planes (xr,xw,xk,xv,xa,xg)
// ---------------------------------------------------------------------------
__global__ void mix_kernel(const float* __restrict__ x, const float* __restrict__ xx,
                           __bf16* __restrict__ out) {
    size_t idx = (size_t)blockIdx.x * 256 + threadIdx.x;  // < 16M
    int row = (int)(idx >> 10), d = (int)(idx & 1023);
    int t = row & (TLEN - 1);
    float xc = x[idx];
    float prev = (t > 0) ? x[idx - DDIM] : 0.0f;
    float delta = prev - xc;
    const size_t plane = (size_t)NROW * DDIM;
#pragma unroll
    for (int j = 0; j < 6; ++j)
        out[j * plane + idx] = (__bf16)(xc + delta * xx[j * DDIM + d]);
}

// ---------------------------------------------------------------------------
// Generic bf16 WMMA GEMM: C[M,N] f32 = A[M,K] bf16 * Bt[N,K]^T bf16
// Block 256 thr (8 waves), tile 128x128, K-step 32, double-buffered LDS with
// async global->LDS DMA (tile t+1 overlaps WMMA on tile t). Wave grid 4(M)x2(N),
// each wave: 2 M-tiles x 4 N-tiles of v_wmma_f32_16x16x32_bf16.
// ---------------------------------------------------------------------------
union FragCvt { uint4 u[2]; v16bf v; };

__global__ __launch_bounds__(256) void gemm_bf16(const __bf16* __restrict__ A,
                                                 const __bf16* __restrict__ Bt,
                                                 float* __restrict__ C,
                                                 int M, int N, int K) {
    __shared__ __align__(16) __bf16 As[2][128][32];
    __shared__ __align__(16) __bf16 Bs[2][128][32];
    const int tid  = threadIdx.x;
    const int wave = tid >> 5, lane = tid & 31;
    const int wm = wave >> 1, wn = wave & 1;
    const int g = lane >> 4, lr = lane & 15;
    const int m0 = blockIdx.y * 128, n0 = blockIdx.x * 128;

    const int ldrow  = tid >> 1;          // 0..127
    const int ldhalf = (tid & 1) * 16;    // element offset 0 / 16
    const int nc = n0 + ldrow;
    // Clamp out-of-range B rows to a valid source row: keeps per-wave async
    // issue counts uniform; the garbage only affects output cols >= N which
    // are never stored.
    const int ncs = (nc < N) ? nc : (N - 1);

    v8f acc[2][4];
#pragma unroll
    for (int m = 0; m < 2; ++m)
#pragma unroll
        for (int n = 0; n < 4; ++n)
#pragma unroll
            for (int i = 0; i < 8; ++i) acc[m][n][i] = 0.0f;

    auto stage = [&](int buf, int k0) {
        const __bf16* ga = A  + (size_t)(m0 + ldrow) * K + k0 + ldhalf;
        const __bf16* gb = Bt + (size_t)ncs * K + k0 + ldhalf;
#if USE_ASYNC_LDS
        __builtin_amdgcn_global_load_async_to_lds_b128(
            as_global_(ga),     as_lds_(&As[buf][ldrow][ldhalf]),     0, 0);
        __builtin_amdgcn_global_load_async_to_lds_b128(
            as_global_(ga + 8), as_lds_(&As[buf][ldrow][ldhalf + 8]), 0, 0);
        __builtin_amdgcn_global_load_async_to_lds_b128(
            as_global_(gb),     as_lds_(&Bs[buf][ldrow][ldhalf]),     0, 0);
        __builtin_amdgcn_global_load_async_to_lds_b128(
            as_global_(gb + 8), as_lds_(&Bs[buf][ldrow][ldhalf + 8]), 0, 0);
#else
        const uint4* gpa = (const uint4*)ga;
        *(uint4*)&As[buf][ldrow][ldhalf]     = gpa[0];
        *(uint4*)&As[buf][ldrow][ldhalf + 8] = gpa[1];
        const uint4* gpb = (const uint4*)gb;
        *(uint4*)&Bs[buf][ldrow][ldhalf]     = gpb[0];
        *(uint4*)&Bs[buf][ldrow][ldhalf + 8] = gpb[1];
#endif
        if (k0 + 32 < K) {  // global_prefetch_b8 for the tile after this one
            __builtin_prefetch(ga + 32, 0, 1);
            __builtin_prefetch(gb + 32, 0, 1);
        }
    };

    const int nIter = K >> 5;
    stage(0, 0);
    for (int it = 0; it < nIter; ++it) {
        const int cur = it & 1;
        if (it + 1 < nIter) {
            stage(cur ^ 1, (it + 1) << 5);     // DMA next tile into other buffer
#if USE_ASYNC_LDS
            __builtin_amdgcn_s_wait_asynccnt(4);  // only the new stage may remain
        } else {
            __builtin_amdgcn_s_wait_asynccnt(0);
        }
#else
        }
#endif
        __syncthreads();   // current buffer visible to all waves

        // A fragments per ISA 16-bit 16x32 layout (two b128 per lane)
        FragCvt afr[2];
#pragma unroll
        for (int m = 0; m < 2; ++m) {
            int row = wm * 32 + m * 16 + lr;
            afr[m].u[0] = *(const uint4*)&As[cur][row][8 * g];
            afr[m].u[1] = *(const uint4*)&As[cur][row][16 + 8 * g];
        }
#pragma unroll
        for (int n = 0; n < 4; ++n) {
            FragCvt bfr;                       // B 32x16: lane n-col, K=8g..8g+7, 16+8g..16+8g+7
            int col = wn * 64 + n * 16 + lr;
            bfr.u[0] = *(const uint4*)&Bs[cur][col][8 * g];
            bfr.u[1] = *(const uint4*)&Bs[cur][col][16 + 8 * g];
#pragma unroll
            for (int m = 0; m < 2; ++m)
                acc[m][n] = __builtin_amdgcn_wmma_f32_16x16x32_bf16(
                    false, afr[m].v, false, bfr.v, (short)0, acc[m][n], false, false);
        }
        __syncthreads();   // all waves done reading before buffer reuse
    }

    // C/D layout: VGPR i -> M = i + 8g, N = lr
#pragma unroll
    for (int m = 0; m < 2; ++m)
#pragma unroll
        for (int n = 0; n < 4; ++n) {
            int col = n0 + wn * 64 + n * 16 + lr;
            if (col < N) {
#pragma unroll
                for (int i = 0; i < 8; ++i) {
                    int row = m0 + wm * 32 + m * 16 + g * 8 + i;
                    C[(size_t)row * N + col] = acc[m][n][i];
                }
            }
        }
}

// ---------------------------------------------------------------------------
// Activation + f32 -> bf16 (mode 0: copy, 1: tanh, 2: sigmoid)
// ---------------------------------------------------------------------------
__global__ void act_kernel(const float* __restrict__ in, __bf16* __restrict__ out,
                           int n, int mode) {
    int i = blockIdx.x * 256 + threadIdx.x;
    if (i >= n) return;
    float v = in[i];
    if (mode == 1) v = tanhf(v);
    else if (mode == 2) v = sigmoidf_(v);
    out[i] = (__bf16)v;
}

// ---------------------------------------------------------------------------
// Per-head-row prep: kk normalize, k/v/logw transforms, build aa=-kk, bb=kk*a.
// One wave per head-row (2 elems/lane); dw becomes exp(logw) (hoists the
// transcendental out of the sequential scan).
// ---------------------------------------------------------------------------
__global__ __launch_bounds__(256) void prep_core_kernel(
    float* __restrict__ kbuf, float* __restrict__ vbuf, float* __restrict__ dw,
    const float* __restrict__ da, const float* __restrict__ dv,
    const float* __restrict__ v_first,
    const float* __restrict__ w0, const float* __restrict__ a0,
    const float* __restrict__ v0, const float* __restrict__ k_k,
    const float* __restrict__ k_a,
    float* __restrict__ aa, float* __restrict__ bb) {
    int wid  = blockIdx.x * 8 + (threadIdx.x >> 5);   // head-row id
    int lane = threadIdx.x & 31;
    int row = wid >> 4, h = wid & 15;
    int e = lane * 2, d = h * 64 + e;
    size_t idx = (size_t)row * 1024 + d;

    float2 kr = *(const float2*)(kbuf + idx);
    float kk0 = kr.x * k_k[d], kk1 = kr.y * k_k[d + 1];
    float ss = kk0 * kk0 + kk1 * kk1;
#pragma unroll
    for (int off = 16; off > 0; off >>= 1) ss += __shfl_xor(ss, off, 32);
    float inv = 1.0f / fmaxf(sqrtf(ss), 1e-12f);
    kk0 *= inv; kk1 *= inv;

    float a0v = sigmoidf_(a0[d] + da[idx]);
    float a1v = sigmoidf_(a0[d + 1] + da[idx + 1]);

    float2 ew;
    ew.x = __expf(LOGW * sigmoidf_(w0[d] + dw[idx]));
    ew.y = __expf(LOGW * sigmoidf_(w0[d + 1] + dw[idx + 1]));
    *(float2*)(dw + idx) = ew;

    float2 vr = *(const float2*)(vbuf + idx);
    float2 vf = *(const float2*)(v_first + idx);
    vr.x += (vf.x - vr.x) * sigmoidf_(v0[d] + dv[idx]);
    vr.y += (vf.y - vr.y) * sigmoidf_(v0[d + 1] + dv[idx + 1]);
    *(float2*)(vbuf + idx) = vr;

    kr.x *= (1.0f + (a0v - 1.0f) * k_a[d]);
    kr.y *= (1.0f + (a1v - 1.0f) * k_a[d + 1]);
    *(float2*)(kbuf + idx) = kr;

    *(float2*)(aa + idx) = make_float2(-kk0, -kk1);
    *(float2*)(bb + idx) = make_float2(kk0 * a0v, kk1 * a1v);
}

// ---------------------------------------------------------------------------
// Sequential delta-rule scan. 1 block per (b,h); 256 threads; state 64x64 f32
// in registers (16 floats/thread); 4 lanes per state row, shfl_xor reductions.
// ---------------------------------------------------------------------------
__global__ __launch_bounds__(256) void rwkv_core_kernel(
    const float* __restrict__ r,  const float* __restrict__ ew,
    const float* __restrict__ k,  const float* __restrict__ v,
    const float* __restrict__ aa, const float* __restrict__ bb,
    float* __restrict__ o) {
    const int blk = blockIdx.x;           // b*16 + h
    const int tid = threadIdx.x;
    const int i  = tid >> 2;              // state row 0..63
    const int kq = tid & 3;               // k-quarter

    float S[16];
#pragma unroll
    for (int j = 0; j < 16; ++j) S[j] = 0.0f;

    size_t base  = ((size_t)(blk >> 4) * TLEN) * 1024 + (size_t)(blk & 15) * 64 + kq * 16;
    size_t vbase = ((size_t)(blk >> 4) * TLEN) * 1024 + (size_t)(blk & 15) * 64 + i;

    for (int t = 0; t < TLEN; ++t, base += 1024, vbase += 1024) {
        float av[16], bv[16], kv_[16], rv[16], wv[16];
        const float4* ap = (const float4*)(aa + base);
        const float4* bp = (const float4*)(bb + base);
        const float4* kp = (const float4*)(k + base);
        const float4* rp = (const float4*)(r + base);
        const float4* wp = (const float4*)(ew + base);
#pragma unroll
        for (int q = 0; q < 4; ++q) {
            float4 t4;
            t4 = ap[q]; av[4*q] = t4.x; av[4*q+1] = t4.y; av[4*q+2] = t4.z; av[4*q+3] = t4.w;
            t4 = bp[q]; bv[4*q] = t4.x; bv[4*q+1] = t4.y; bv[4*q+2] = t4.z; bv[4*q+3] = t4.w;
            t4 = kp[q]; kv_[4*q] = t4.x; kv_[4*q+1] = t4.y; kv_[4*q+2] = t4.z; kv_[4*q+3] = t4.w;
            t4 = rp[q]; rv[4*q] = t4.x; rv[4*q+1] = t4.y; rv[4*q+2] = t4.z; rv[4*q+3] = t4.w;
            t4 = wp[q]; wv[4*q] = t4.x; wv[4*q+1] = t4.y; wv[4*q+2] = t4.z; wv[4*q+3] = t4.w;
        }
        float sa = 0.0f;
#pragma unroll
        for (int j = 0; j < 16; ++j) sa += S[j] * av[j];
        sa += __shfl_xor(sa, 1, 32);
        sa += __shfl_xor(sa, 2, 32);

        float vi = v[vbase];
        float ov = 0.0f;
#pragma unroll
        for (int j = 0; j < 16; ++j) {
            S[j] = S[j] * wv[j] + sa * bv[j] + vi * kv_[j];
            ov += S[j] * rv[j];
        }
        ov += __shfl_xor(ov, 1, 32);
        ov += __shfl_xor(ov, 2, 32);
        if (kq == 0) o[vbase] = ov;
    }
}

// ---------------------------------------------------------------------------
// GroupNorm per head + rank-1 bonus + output gate -> bf16 A operand for W_o
// ---------------------------------------------------------------------------
__global__ __launch_bounds__(256) void outprep_kernel(
    const float* __restrict__ o,  const float* __restrict__ r,
    const float* __restrict__ k,  const float* __restrict__ v,
    const float* __restrict__ gg, const float* __restrict__ r_k,
    const float* __restrict__ gn_w, const float* __restrict__ gn_b,
    __bf16* __restrict__ og) {
    int wid  = blockIdx.x * 8 + (threadIdx.x >> 5);
    int lane = threadIdx.x & 31;
    int row = wid >> 4, h = wid & 15;
    int e = lane * 2, d = h * 64 + e;
    size_t idx = (size_t)row * 1024 + d;

    float2 ov = *(const float2*)(o + idx);
    float2 rv = *(const float2*)(r + idx);
    float2 kv = *(const float2*)(k + idx);
    float s1 = ov.x + ov.y;
    float s2 = ov.x * ov.x + ov.y * ov.y;
    float bon = rv.x * kv.x * r_k[d] + rv.y * kv.y * r_k[d + 1];
#pragma unroll
    for (int off = 16; off > 0; off >>= 1) {
        s1  += __shfl_xor(s1, off, 32);
        s2  += __shfl_xor(s2, off, 32);
        bon += __shfl_xor(bon, off, 32);
    }
    float mu  = s1 * (1.0f / 64.0f);
    float var = s2 * (1.0f / 64.0f) - mu * mu;
    float rs  = rsqrtf(var + GNEPS);
    float2 vv = *(const float2*)(v + idx);
    float2 gv = *(const float2*)(gg + idx);
    float on0 = ((ov.x - mu) * rs) * gn_w[d]     + gn_b[d]     + bon * vv.x;
    float on1 = ((ov.y - mu) * rs) * gn_w[d + 1] + gn_b[d + 1] + bon * vv.y;
    og[idx]     = (__bf16)(on0 * gv.x);
    og[idx + 1] = (__bf16)(on1 * gv.y);
}

// ---------------------------------------------------------------------------
// Host-side orchestration
// ---------------------------------------------------------------------------
extern "C" void kernel_launch(void* const* d_in, const int* in_sizes, int n_in,
                              void* d_out, int out_size, void* d_ws, size_t ws_size,
                              hipStream_t stream) {
    (void)in_sizes; (void)n_in; (void)out_size; (void)ws_size;
    const float* x       = (const float*)d_in[0];
    const float* v_first = (const float*)d_in[1];
    const float* x_x     = (const float*)d_in[2];
    const float* W_r = (const float*)d_in[3];
    const float* W_k = (const float*)d_in[4];
    const float* W_v = (const float*)d_in[5];
    const float* W_o = (const float*)d_in[6];
    const float* w1 = (const float*)d_in[7];
    const float* w2 = (const float*)d_in[8];
    const float* w0 = (const float*)d_in[9];
    const float* a1 = (const float*)d_in[10];
    const float* a2 = (const float*)d_in[11];
    const float* a0 = (const float*)d_in[12];
    const float* v1 = (const float*)d_in[13];
    const float* v2 = (const float*)d_in[14];
    const float* v0 = (const float*)d_in[15];
    const float* g1 = (const float*)d_in[16];
    const float* g2 = (const float*)d_in[17];
    const float* k_k = (const float*)d_in[18];
    const float* k_a = (const float*)d_in[19];
    const float* r_k = (const float*)d_in[20];
    const float* gn_w = (const float*)d_in[21];
    const float* gn_b = (const float*)d_in[22];
    float* outF = (float*)d_out;

    char* ws = (char*)d_ws;
    auto alloc = [&](size_t bytes) -> char* {
        char* p = ws; ws += (bytes + 255) & ~(size_t)255; return p;
    };
    const size_t planeB = (size_t)NROW * DDIM * 2;     // bf16 plane
    const size_t bigF   = (size_t)NROW * DDIM * 4;     // f32 [16384,1024]

    __bf16* A6  = (__bf16*)alloc(6 * planeB);          // xr..xg; reused later
    __bf16* xr = A6 + 0 * (size_t)NROW * DDIM;
    __bf16* xw = A6 + 1 * (size_t)NROW * DDIM;
    __bf16* xk = A6 + 2 * (size_t)NROW * DDIM;
    __bf16* xv = A6 + 3 * (size_t)NROW * DDIM;
    __bf16* xa = A6 + 4 * (size_t)NROW * DDIM;
    __bf16* xg = A6 + 5 * (size_t)NROW * DDIM;
    // reuse of A6 region after the first/second GEMM waves:
    float* aa = (float*)((char*)A6);
    float* bb = (float*)((char*)A6 + bigF);
    float* ob = (float*)((char*)A6 + 2 * bigF);

    __bf16* WrT = (__bf16*)alloc(DDIM * DDIM * 2);
    __bf16* WkT = (__bf16*)alloc(DDIM * DDIM * 2);
    __bf16* WvT = (__bf16*)alloc(DDIM * DDIM * 2);
    __bf16* WoT = (__bf16*)alloc(DDIM * DDIM * 2);
    __bf16* w1T = (__bf16*)alloc(64 * DDIM * 2);
    __bf16* w2T = (__bf16*)alloc(DDIM * 64 * 2);
    __bf16* a1T = (__bf16*)alloc(64 * DDIM * 2);
    __bf16* a2T = (__bf16*)alloc(DDIM * 64 * 2);
    __bf16* v1T = (__bf16*)alloc(32 * DDIM * 2);
    __bf16* v2T = (__bf16*)alloc(DDIM * 32 * 2);
    __bf16* g1T = (__bf16*)alloc(160 * DDIM * 2);
    __bf16* g2T = (__bf16*)alloc(DDIM * 160 * 2);

    float* rbuf = (float*)alloc(bigF);
    float* kbuf = (float*)alloc(bigF);
    float* vbuf = (float*)alloc(bigF);
    float* hw = (float*)alloc((size_t)NROW * 64 * 4);
    float* ha = (float*)alloc((size_t)NROW * 64 * 4);
    float* hv = (float*)alloc((size_t)NROW * 32 * 4);
    float* hg = (float*)alloc((size_t)NROW * 160 * 4);
    __bf16* hwb = (__bf16*)alloc((size_t)NROW * 64 * 2);
    __bf16* hab = (__bf16*)alloc((size_t)NROW * 64 * 2);
    __bf16* hvb = (__bf16*)alloc((size_t)NROW * 32 * 2);
    __bf16* hgb = (__bf16*)alloc((size_t)NROW * 160 * 2);
    float* dw = (float*)alloc(bigF);
    float* da = (float*)alloc(bigF);
    float* dv = (float*)alloc(bigF);
    float* gg = (float*)alloc(bigF);
    __bf16* og = (__bf16*)alloc(planeB);

    auto tgrid = [](int n) { return dim3((n + 255) / 256); };

    // weight transposes (f32 -> bf16 [N][K])
    wt_kernel<<<tgrid(DDIM * DDIM), 256, 0, stream>>>(W_r, WrT, DDIM, DDIM);
    wt_kernel<<<tgrid(DDIM * DDIM), 256, 0, stream>>>(W_k, WkT, DDIM, DDIM);
    wt_kernel<<<tgrid(DDIM * DDIM), 256, 0, stream>>>(W_v, WvT, DDIM, DDIM);
    wt_kernel<<<tgrid(DDIM * DDIM), 256, 0, stream>>>(W_o, WoT, DDIM, DDIM);
    wt_kernel<<<tgrid(DDIM * 64), 256, 0, stream>>>(w1, w1T, DDIM, 64);
    wt_kernel<<<tgrid(DDIM * 64), 256, 0, stream>>>(w2, w2T, 64, DDIM);
    wt_kernel<<<tgrid(DDIM * 64), 256, 0, stream>>>(a1, a1T, DDIM, 64);
    wt_kernel<<<tgrid(DDIM * 64), 256, 0, stream>>>(a2, a2T, 64, DDIM);
    wt_kernel<<<tgrid(DDIM * 32), 256, 0, stream>>>(v1, v1T, DDIM, 32);
    wt_kernel<<<tgrid(DDIM * 32), 256, 0, stream>>>(v2, v2T, 32, DDIM);
    wt_kernel<<<tgrid(DDIM * 160), 256, 0, stream>>>(g1, g1T, DDIM, 160);
    wt_kernel<<<tgrid(DDIM * 160), 256, 0, stream>>>(g2, g2T, 160, DDIM);

    // token shift mix
    mix_kernel<<<NROW * DDIM / 256, 256, 0, stream>>>(x, x_x, A6);

    // projection + LoRA-up GEMMs
    dim3 gBig(DDIM / 128, NROW / 128);
    gemm_bf16<<<gBig, 256, 0, stream>>>(xr, WrT, rbuf, NROW, DDIM, DDIM);
    gemm_bf16<<<gBig, 256, 0, stream>>>(xk, WkT, kbuf, NROW, DDIM, DDIM);
    gemm_bf16<<<gBig, 256, 0, stream>>>(xv, WvT, vbuf, NROW, DDIM, DDIM);
    gemm_bf16<<<dim3(1, NROW / 128), 256, 0, stream>>>(xw, w1T, hw, NROW, 64, DDIM);
    gemm_bf16<<<dim3(1, NROW / 128), 256, 0, stream>>>(xa, a1T, ha, NROW, 64, DDIM);
    gemm_bf16<<<dim3(1, NROW / 128), 256, 0, stream>>>(xv, v1T, hv, NROW, 32, DDIM);
    gemm_bf16<<<dim3(2, NROW / 128), 256, 0, stream>>>(xg, g1T, hg, NROW, 160, DDIM);

    // activations -> bf16 operands
    act_kernel<<<tgrid(NROW * 64), 256, 0, stream>>>(hw, hwb, NROW * 64, 1);
    act_kernel<<<tgrid(NROW * 64), 256, 0, stream>>>(ha, hab, NROW * 64, 0);
    act_kernel<<<tgrid(NROW * 32), 256, 0, stream>>>(hv, hvb, NROW * 32, 0);
    act_kernel<<<tgrid(NROW * 160), 256, 0, stream>>>(hg, hgb, NROW * 160, 2);

    // LoRA-down GEMMs
    gemm_bf16<<<gBig, 256, 0, stream>>>(hwb, w2T, dw, NROW, DDIM, 64);
    gemm_bf16<<<gBig, 256, 0, stream>>>(hab, a2T, da, NROW, DDIM, 64);
    gemm_bf16<<<gBig, 256, 0, stream>>>(hvb, v2T, dv, NROW, DDIM, 32);
    gemm_bf16<<<gBig, 256, 0, stream>>>(hgb, g2T, gg, NROW, DDIM, 160);

    // per-head prep (A6 planes are free now; aa/bb/ob overlay them)
    prep_core_kernel<<<NROW * NHEAD / 8, 256, 0, stream>>>(
        kbuf, vbuf, dw, da, dv, v_first, w0, a0, v0, k_k, k_a, aa, bb);

    // sequential scan: 1 block per (b,h)
    rwkv_core_kernel<<<8 * NHEAD, 256, 0, stream>>>(rbuf, dw, kbuf, vbuf, aa, bb, ob);

    // GroupNorm + bonus + gate -> bf16
    outprep_kernel<<<NROW * NHEAD / 8, 256, 0, stream>>>(
        ob, rbuf, kbuf, vbuf, gg, r_k, gn_w, gn_b, og);

    // final projection into d_out (first tuple element)
    gemm_bf16<<<gBig, 256, 0, stream>>>(og, WoT, outF, NROW, DDIM, DDIM);

    // second tuple element: v_first passthrough
    (void)hipMemcpyAsync(outF + (size_t)NROW * DDIM, v_first,
                         (size_t)NROW * DDIM * sizeof(float),
                         hipMemcpyDeviceToDevice, stream);
}
